// Seq2Seq_37692632990334
// MI455X (gfx1250) — matrix operand order repaired and checked by reading, not compiled, not run on hardware
//
#include <hip/hip_runtime.h>
#include <hip/hip_bf16.h>
#include <math.h>

// ---------------------------------------------------------------------------
// Seq2Seq forward (encoder 2-layer bi-LSTM + attention decoder + NLL loss)
// for MI455X / gfx1250.
//
// All matmuls go through v_wmma_f32_16x16x32_f16 (f16 in, f32 accumulate).
// Weights are converted once per launch into a WMMA-B-fragment-swizzled f16
// layout: tile (kt,nt) of B = W^T stored as [lane 0..31][16 halves], so each
// lane's fragment is one contiguous 32-byte load. A fragments load straight
// from row-major f16 activations (two b128 loads per lane per K-tile).
// Each wave computes a 2x2 block of 16x16 C tiles (32x32 of C): 8 b128 loads
// feed 4 WMMAs per K-tile, doubling wmma:vmem density vs 1-tile waves.
// Requires M % 32 == 0 and N % 32 == 0 (true for every GEMM here).
// ---------------------------------------------------------------------------

typedef _Float16 half_t;
typedef _Float16 v8h  __attribute__((ext_vector_type(8)));
typedef _Float16 v16h __attribute__((ext_vector_type(16)));
typedef float    v8f  __attribute__((ext_vector_type(8)));

#define B_    32
#define TS_   64
#define TT_   66
#define STEPS 64
#define E_    512
#define H_    512
#define V_    16000
#define NEGI  (-1e9f)

__device__ __forceinline__ float sigm(float x) { return 1.0f / (1.0f + __expf(-x)); }

// ---------------------------------------------------------------------------
// Generic tiled GEMM: C[M,N] (f32) = A[M,K](f16, row-major, lda) x Bswz + add.
// Bswz: swizzled fragments, tile (kt,nt) at ((kt*nTiles+nt)*512), lane*16.
// addMode: 0 = none, 1 = bias[N], 2 = full matrix addend[M,ldc].
// act: 1 -> tanh on output. Dh (optional): f16 copy of output, stride ldh.
// One wave computes a 2x2 block of 16x16 C tiles, looping over kTiles.
// ---------------------------------------------------------------------------
__global__ void k_gemm(const half_t* __restrict__ A, int lda,
                       const half_t* __restrict__ Bswz,
                       float* __restrict__ C, int ldc,
                       const float* __restrict__ addend, int addMode,
                       half_t* __restrict__ Dh, int ldh, int act,
                       int mTiles, int nTiles, int kTiles)
{
    int wave = (int)((blockIdx.x * blockDim.x + threadIdx.x) >> 5);
    int lane = (int)(threadIdx.x & 31);
    int nw   = (int)((gridDim.x * blockDim.x) >> 5);
    int mB   = mTiles >> 1;
    int nB   = nTiles >> 1;
    int total = mB * nB;
    int m16  = lane & 15;
    int hi   = lane >> 4;

    for (int blk = wave; blk < total; blk += nw) {
        int mb = blk / nB;
        int nb = blk - mb * nB;
        int mt0 = mb * 2, nt0 = nb * 2;

        v8f acc00 = {}, acc01 = {}, acc10 = {}, acc11 = {};
        const half_t* arow0 = A + (size_t)(mt0 * 16 + m16) * lda + 8 * hi;
        const half_t* arow1 = arow0 + (size_t)16 * lda;
        const half_t* bp    = Bswz + ((size_t)nt0 << 9) + ((size_t)lane << 4);
        size_t bstride = (size_t)nTiles << 9;

        for (int kt = 0; kt < kTiles; ++kt) {
            union { v16h v; v8h h[2]; } a0, a1, b0, b1;
            a0.h[0] = *(const v8h*)(arow0);
            a0.h[1] = *(const v8h*)(arow0 + 16);
            a1.h[0] = *(const v8h*)(arow1);
            a1.h[1] = *(const v8h*)(arow1 + 16);
            b0.h[0] = *(const v8h*)(bp);
            b0.h[1] = *(const v8h*)(bp + 8);
            b1.h[0] = *(const v8h*)(bp + 512);
            b1.h[1] = *(const v8h*)(bp + 520);
            if (kt + 1 < kTiles) __builtin_prefetch(bp + bstride, 0, 1);
            acc00 = __builtin_amdgcn_wmma_f32_16x16x32_f16(
                        false, a0.v, false, b0.v, (short)0, acc00, false, false);
            acc01 = __builtin_amdgcn_wmma_f32_16x16x32_f16(
                        false, a0.v, false, b1.v, (short)0, acc01, false, false);
            acc10 = __builtin_amdgcn_wmma_f32_16x16x32_f16(
                        false, a1.v, false, b0.v, (short)0, acc10, false, false);
            acc11 = __builtin_amdgcn_wmma_f32_16x16x32_f16(
                        false, a1.v, false, b1.v, (short)0, acc11, false, false);
            arow0 += 32;
            arow1 += 32;
            bp    += bstride;
        }

        #pragma unroll
        for (int q = 0; q < 4; ++q) {
            const v8f* accp = (q == 0) ? &acc00 : (q == 1) ? &acc01
                            : (q == 2) ? &acc10 : &acc11;
            int mt = mt0 + (q >> 1);
            int nt = nt0 + (q & 1);
            int col = nt * 16 + m16;
            #pragma unroll
            for (int r = 0; r < 8; ++r) {
                int row = mt * 16 + r + 8 * hi;
                float v = (*accp)[r];
                if (addMode == 1)      v += addend[col];
                else if (addMode == 2) v += addend[(size_t)row * ldc + col];
                if (act) v = tanhf(v);
                C[(size_t)row * ldc + col] = v;
                if (Dh) Dh[(size_t)row * ldh + col] = (half_t)v;
            }
        }
    }
}

// Convert f32 weight W[Nreal,K] (torch [out,in]) into WMMA B-fragment layout
// for computing x @ W^T. B = W^T (K x N). Element idx of lane's v16h maps to
// K = 8*(lane>>4) + (idx&7) + 16*(idx>>3), N = nt*16 + (lane&15). Zero-pads n >= Nreal.
__global__ void k_swizzle(const float* __restrict__ W, int Nreal, int K,
                          half_t* __restrict__ out, int nTiles, int kTiles)
{
    size_t id  = (size_t)blockIdx.x * blockDim.x + threadIdx.x;
    size_t tot = (size_t)kTiles * nTiles * 512;
    if (id >= tot) return;
    int idx  = (int)(id & 15);
    int lane = (int)((id >> 4) & 31);
    size_t tile = id >> 9;
    int nt = (int)(tile % nTiles);
    int kt = (int)(tile / nTiles);
    int n = nt * 16 + (lane & 15);
    int k = kt * 32 + 8 * (lane >> 4) + (idx & 7) + 16 * (idx >> 3);
    float v = (n < Nreal) ? W[(size_t)n * K + k] : 0.0f;
    out[id] = (half_t)v;
}

// Embedding gather -> f16 activations, rows laid out [t*B + b][E].
__global__ void k_embed(const int* __restrict__ ids, int idsLd,
                        const float* __restrict__ emb,
                        half_t* __restrict__ out, int B, int T, int E)
{
    size_t id  = (size_t)blockIdx.x * blockDim.x + threadIdx.x;
    size_t tot = (size_t)B * T * E;
    if (id >= tot) return;
    int e = (int)(id % E);
    size_t row = id / E;
    int b = (int)(row % B);
    int t = (int)(row / B);
    int tok = ids[(size_t)b * idsLd + t];
    out[id] = (half_t)emb[(size_t)tok * E + e];
}

__global__ void k_zero32(unsigned int* __restrict__ p, size_t n)
{
    size_t id = (size_t)blockIdx.x * blockDim.x + threadIdx.x;
    if (id < n) p[id] = 0u;
}

// Encoder LSTM cell (H=512), both directions in one launch.
// gates: [2][B][2048]. States: [2][B][512] (h f32, c f32, h f16).
// xout (optional): f16 dest rows (tt*B+b), col dir*512  (layer-1 input).
// encout (optional): f32 dest [B][TS][1024] at col dir*512 (attention keys).
__global__ void k_lstm_step(const float* __restrict__ gates,
                            float* __restrict__ h, float* __restrict__ c,
                            half_t* __restrict__ hh,
                            half_t* __restrict__ xout, int xldc,
                            float* __restrict__ encout,
                            int tfwd, int tbwd)
{
    int id = blockIdx.x * blockDim.x + threadIdx.x;
    if (id >= 2 * B_ * H_) return;
    int j   = id % H_;
    int r   = id / H_;
    int b   = r % B_;
    int dir = r / B_;
    const float* g = gates + (size_t)dir * B_ * 2048 + (size_t)b * 2048;
    float iv = g[j], fv = g[H_ + j], gv = g[2 * H_ + j], ov = g[3 * H_ + j];
    size_t si = (size_t)dir * B_ * H_ + (size_t)b * H_ + j;
    float cc = sigm(fv) * c[si] + sigm(iv) * tanhf(gv);
    float hv = sigm(ov) * tanhf(cc);
    c[si] = cc; h[si] = hv; hh[si] = (half_t)hv;
    int tt = dir ? tbwd : tfwd;
    if (xout)   xout[((size_t)tt * B_ + b) * xldc + dir * H_ + j] = (half_t)hv;
    if (encout) encout[((size_t)b * TS_ + tt) * 1024 + dir * H_ + j] = hv;
}

// Decoder LSTM cell (hidden 2H = 1024) + target-PAD masking of h.
__global__ void k_dec_cell(const float* __restrict__ gates,
                           const int* __restrict__ trg, int s,
                           float* __restrict__ h, float* __restrict__ c,
                           half_t* __restrict__ hh)
{
    int id = blockIdx.x * blockDim.x + threadIdx.x;
    if (id >= B_ * 1024) return;
    int j = id % 1024;
    int b = id / 1024;
    const float* g = gates + (size_t)b * 4096;
    float iv = g[j], fv = g[1024 + j], gv = g[2048 + j], ov = g[3072 + j];
    size_t si = (size_t)b * 1024 + j;
    float cc = sigm(fv) * c[si] + sigm(iv) * tanhf(gv);
    float hv = sigm(ov) * tanhf(cc);
    int tgt = trg[(size_t)b * TT_ + s + 1];
    if (tgt == 0) hv = 0.0f;            // decoder_hidden_t[mask] = 0 (PAD)
    c[si] = cc; h[si] = hv; hh[si] = (half_t)hv;
}

// Build decoder initial state from encoder layer-0 final fwd/bwd states.
__global__ void k_init_dec(const float* __restrict__ h0, const float* __restrict__ c0,
                           float* __restrict__ dh, float* __restrict__ dc,
                           half_t* __restrict__ dhh)
{
    int id = blockIdx.x * blockDim.x + threadIdx.x;
    if (id >= B_ * 1024) return;
    int j = id % 1024;
    int b = id / 1024;
    int dir = j >> 9;
    int jj  = j & 511;
    size_t si = (size_t)dir * B_ * H_ + (size_t)b * H_ + jj;
    float hv = h0[si], cv = c0[si];
    dh[id] = hv; dc[id] = cv; dhh[id] = (half_t)hv;
}

// Dot attention + context + f16 concat([h, ctx]) build. One block per batch b.
__global__ void k_attn(const float* __restrict__ encout,  // [B][TS][1024]
                       const float* __restrict__ h,       // [B][1024]
                       const int* __restrict__ src,       // [B][TS] ids (PAD mask)
                       half_t* __restrict__ concat)       // [B][2048]
{
    __shared__ float sh[1024];
    __shared__ float sw[TS_];
    int b = blockIdx.x;
    int tid = threadIdx.x;
    int warp = tid >> 5, lane = tid & 31;

    for (int k = tid; k < 1024; k += 256) sh[k] = h[(size_t)b * 1024 + k];
    __syncthreads();

    // scores: warp w covers t = 8w .. 8w+7, wave32 reduction per score
    for (int t0 = 0; t0 < 8; ++t0) {
        int t = warp * 8 + t0;
        const float* e = encout + ((size_t)b * TS_ + t) * 1024;
        float s = 0.0f;
        for (int k = lane; k < 1024; k += 32) s += e[k] * sh[k];
        for (int off = 16; off; off >>= 1) s += __shfl_xor(s, off, 32);
        if (lane == 0) {
            if (src[(size_t)b * TS_ + t] == 0) s = NEGI;
            sw[t] = s;
        }
    }
    __syncthreads();

    // softmax over 64 scores (warp 0, two per lane)
    if (warp == 0) {
        float v0 = sw[lane], v1 = sw[32 + lane];
        float mm = fmaxf(v0, v1);
        for (int off = 16; off; off >>= 1) mm = fmaxf(mm, __shfl_xor(mm, off, 32));
        float e0 = __expf(v0 - mm), e1 = __expf(v1 - mm);
        float ss = e0 + e1;
        for (int off = 16; off; off >>= 1) ss += __shfl_xor(ss, off, 32);
        float inv = 1.0f / ss;
        sw[lane] = e0 * inv; sw[32 + lane] = e1 * inv;
    }
    __syncthreads();

    // context + concat
    for (int k = tid; k < 1024; k += 256) {
        const float* e = encout + (size_t)b * TS_ * 1024 + k;
        float acc = 0.0f;
        #pragma unroll 4
        for (int t = 0; t < TS_; ++t) acc += sw[t] * e[(size_t)t * 1024];
        concat[(size_t)b * 2048 + k]        = (half_t)sh[k];
        concat[(size_t)b * 2048 + 1024 + k] = (half_t)acc;
    }
}

// logsumexp NLL + argmax over the 15996 real logits. One block per batch b.
__global__ void k_loss(const float* __restrict__ logits,  // [B][16000]
                       const int* __restrict__ trg, int s,
                       float* __restrict__ lossAcc,
                       float* __restrict__ preds)          // d_out + 1
{
    __shared__ float smax[256];
    __shared__ int   sidx[256];
    __shared__ float ssum[256];
    int b = blockIdx.x;
    int tid = threadIdx.x;
    const float* lg = logits + (size_t)b * V_;

    float m = -1e30f; int mi = 0;
    for (int n = tid; n < V_ - 4; n += 256) {
        float v = lg[n];
        if (v > m) { m = v; mi = n; }
    }
    smax[tid] = m; sidx[tid] = mi;
    __syncthreads();
    for (int st = 128; st; st >>= 1) {
        if (tid < st && smax[tid + st] > smax[tid]) {
            smax[tid] = smax[tid + st]; sidx[tid] = sidx[tid + st];
        }
        __syncthreads();
    }
    float gm = smax[0];
    int   gi = sidx[0];

    float se = 0.0f;
    for (int n = tid; n < V_ - 4; n += 256) se += __expf(lg[n] - gm);
    ssum[tid] = se;
    __syncthreads();
    for (int st = 128; st; st >>= 1) {
        if (tid < st) ssum[tid] += ssum[tid + st];
        __syncthreads();
    }
    if (tid == 0) {
        float lse = gm + logf(ssum[0]);
        int tgt = trg[(size_t)b * TT_ + s + 1];
        if (tgt != 0) atomicAdd(lossAcc, lse - lg[tgt - 4]);
        preds[(size_t)b * STEPS + s] = (float)(gi + 4);
    }
}

__global__ void k_finalize(const float* __restrict__ lossAcc,
                           const int* __restrict__ trg,
                           float* __restrict__ out0)
{
    __shared__ int cnt[256];
    int tid = threadIdx.x;
    int c = 0;
    for (int i = tid; i < B_ * (TT_ - 2); i += 256) {
        int b = i / (TT_ - 2), t = i % (TT_ - 2);
        if (trg[(size_t)b * TT_ + 1 + t] != 0) c++;
    }
    cnt[tid] = c;
    __syncthreads();
    for (int st = 128; st; st >>= 1) {
        if (tid < st) cnt[tid] += cnt[tid + st];
        __syncthreads();
    }
    if (tid == 0) out0[0] = lossAcc[0] / (float)cnt[0];
}

// ---------------------------------------------------------------------------
// Host side
// ---------------------------------------------------------------------------
static void gemm_launch(hipStream_t st, const half_t* A, int lda, const half_t* Bsw,
                        float* C, int ldc, const float* add, int addMode,
                        half_t* Dh, int ldh, int act, int M, int N, int K)
{
    int mT = M >> 4, nT = N >> 4, kT = K >> 5;
    long blocks2x2 = (long)(mT >> 1) * (nT >> 1);
    long threads = blocks2x2 * 32;
    long grid = (threads + 255) / 256;
    if (grid < 1) grid = 1;
    if (grid > 4096) grid = 4096;
    k_gemm<<<dim3((unsigned)grid), dim3(256), 0, st>>>(
        A, lda, Bsw, C, ldc, add, addMode, Dh, ldh, act, mT, nT, kT);
}

static void swz_launch(hipStream_t st, const float* W, int Nreal, int Npad, int K, half_t* out)
{
    int nT = Npad >> 4, kT = K >> 5;
    size_t tot = (size_t)kT * nT * 512;
    unsigned grid = (unsigned)((tot + 255) / 256);
    k_swizzle<<<dim3(grid), dim3(256), 0, st>>>(W, Nreal, K, out, nT, kT);
}

extern "C" void kernel_launch(void* const* d_in, const int* in_sizes, int n_in,
                              void* d_out, int out_size, void* d_ws, size_t ws_size,
                              hipStream_t stream)
{
    (void)in_sizes; (void)n_in; (void)out_size; (void)ws_size;

    const int*   src      = (const int*)  d_in[0];
    const int*   trg      = (const int*)  d_in[1];
    const float* src_emb  = (const float*)d_in[2];
    const float* trg_emb  = (const float*)d_in[3];
    const float* e0_Wih   = (const float*)d_in[4];
    const float* e0_Whh   = (const float*)d_in[5];
    const float* e0_b     = (const float*)d_in[6];
    const float* e1_Wih   = (const float*)d_in[7];
    const float* e1_Whh   = (const float*)d_in[8];
    const float* e1_b     = (const float*)d_in[9];
    const float* d_Wih    = (const float*)d_in[10];
    const float* d_Whh    = (const float*)d_in[11];
    const float* d_b      = (const float*)d_in[12];
    const float* o_W1     = (const float*)d_in[13];
    const float* o_W2     = (const float*)d_in[14];

    float* out  = (float*)d_out;
    char*  ws   = (char*)d_ws;

    // ---- workspace layout (256B-aligned bump allocator) ----
    size_t off = 0;
    auto alloc = [&](size_t bytes) -> size_t {
        off = (off + 255) & ~(size_t)255;
        size_t o = off; off += bytes; return o;
    };
    const size_t SZ_512x2048  = (size_t)16 * 128 * 512;   // halves per swizzled K512/N2048
    const size_t SZ_1024x2048 = (size_t)32 * 128 * 512;
    const size_t SZ_512x4096  = (size_t)16 * 256 * 512;
    const size_t SZ_1024x4096 = (size_t)32 * 256 * 512;
    const size_t SZ_2048x512  = (size_t)64 * 32  * 512;
    const size_t SZ_512x16000 = (size_t)16 * 1000 * 512;

    half_t* sw_e0_ih = (half_t*)(ws + alloc(2 * SZ_512x2048  * 2));
    half_t* sw_e0_hh = (half_t*)(ws + alloc(2 * SZ_512x2048  * 2));
    half_t* sw_e1_ih = (half_t*)(ws + alloc(2 * SZ_1024x2048 * 2));
    half_t* sw_e1_hh = (half_t*)(ws + alloc(2 * SZ_512x2048  * 2));
    half_t* sw_d_ih  = (half_t*)(ws + alloc(SZ_512x4096  * 2));
    half_t* sw_d_hh  = (half_t*)(ws + alloc(SZ_1024x4096 * 2));
    half_t* sw_o1    = (half_t*)(ws + alloc(SZ_2048x512  * 2));
    half_t* sw_o2    = (half_t*)(ws + alloc(SZ_512x16000 * 2));

    half_t* x_src    = (half_t*)(ws + alloc((size_t)TS_ * B_ * E_ * 2));
    half_t* x_trg    = (half_t*)(ws + alloc((size_t)STEPS * B_ * E_ * 2));
    float*  g0_in    = (float*) (ws + alloc((size_t)2 * TS_ * B_ * 2048 * 4));
    half_t* x1       = (half_t*)(ws + alloc((size_t)TS_ * B_ * 1024 * 2));
    float*  g1_in    = (float*) (ws + alloc((size_t)2 * TS_ * B_ * 2048 * 4));
    float*  gdec_in  = (float*) (ws + alloc((size_t)STEPS * B_ * 4096 * 4));
    float*  gates_e  = (float*) (ws + alloc((size_t)2 * B_ * 2048 * 4));
    float*  gates_d  = (float*) (ws + alloc((size_t)B_ * 4096 * 4));
    float*  enc_out  = (float*) (ws + alloc((size_t)B_ * TS_ * 1024 * 4));
    half_t* concat   = (half_t*)(ws + alloc((size_t)B_ * 2048 * 2));
    float*  zt       = (float*) (ws + alloc((size_t)B_ * 512 * 4));
    half_t* zt_h     = (half_t*)(ws + alloc((size_t)B_ * 512 * 2));
    float*  logits   = (float*) (ws + alloc((size_t)B_ * V_ * 4));
    float*  dec_h    = (float*) (ws + alloc((size_t)B_ * 1024 * 4));
    float*  dec_c    = (float*) (ws + alloc((size_t)B_ * 1024 * 4));
    half_t* dec_hh   = (half_t*)(ws + alloc((size_t)B_ * 1024 * 2));

    // zero-initialized region: encoder states + loss accumulator
    size_t zs = (off + 255) & ~(size_t)255;
    float*  h0   = (float*) (ws + alloc((size_t)2 * B_ * H_ * 4));
    float*  c0   = (float*) (ws + alloc((size_t)2 * B_ * H_ * 4));
    half_t* hh0  = (half_t*)(ws + alloc((size_t)2 * B_ * H_ * 2));
    float*  h1   = (float*) (ws + alloc((size_t)2 * B_ * H_ * 4));
    float*  c1   = (float*) (ws + alloc((size_t)2 * B_ * H_ * 4));
    half_t* hh1  = (half_t*)(ws + alloc((size_t)2 * B_ * H_ * 2));
    float*  lacc = (float*) (ws + alloc(4));
    size_t ze = off;

    // ---- 0: zero states + loss ----
    {
        size_t ndw = (ze - zs + 3) / 4;
        unsigned grid = (unsigned)((ndw + 255) / 256);
        k_zero32<<<dim3(grid), dim3(256), 0, stream>>>((unsigned int*)(ws + zs), ndw);
    }

    // ---- 1: weight conversion + swizzle ----
    for (int d = 0; d < 2; ++d) {
        swz_launch(stream, e0_Wih + (size_t)d * 2048 * 512,  2048, 2048, 512,  sw_e0_ih + d * SZ_512x2048);
        swz_launch(stream, e0_Whh + (size_t)d * 2048 * 512,  2048, 2048, 512,  sw_e0_hh + d * SZ_512x2048);
        swz_launch(stream, e1_Wih + (size_t)d * 2048 * 1024, 2048, 2048, 1024, sw_e1_ih + d * SZ_1024x2048);
        swz_launch(stream, e1_Whh + (size_t)d * 2048 * 512,  2048, 2048, 512,  sw_e1_hh + d * SZ_512x2048);
    }
    swz_launch(stream, d_Wih, 4096, 4096, 512,  sw_d_ih);
    swz_launch(stream, d_Whh, 4096, 4096, 1024, sw_d_hh);
    swz_launch(stream, o_W1,  512,  512,  2048, sw_o1);
    swz_launch(stream, o_W2,  V_ - 4, V_, 512,  sw_o2);

    // ---- 2: embeddings ----
    {
        size_t tot = (size_t)TS_ * B_ * E_;
        k_embed<<<dim3((unsigned)((tot + 255) / 256)), dim3(256), 0, stream>>>(
            src, TS_, src_emb, x_src, B_, TS_, E_);
        tot = (size_t)STEPS * B_ * E_;
        k_embed<<<dim3((unsigned)((tot + 255) / 256)), dim3(256), 0, stream>>>(
            trg, TT_, trg_emb, x_trg, B_, STEPS, E_);
    }

    // ---- 3: encoder layer 0 input-gate precompute: g0 = x @ Wih^T + b ----
    for (int d = 0; d < 2; ++d)
        gemm_launch(stream, x_src, E_, sw_e0_ih + d * SZ_512x2048,
                    g0_in + (size_t)d * TS_ * B_ * 2048, 2048,
                    e0_b + (size_t)d * 2048, 1, nullptr, 0, 0,
                    TS_ * B_, 2048, E_);

    // ---- 4: encoder layer 0 recurrence (fwd + bwd per step) ----
    for (int t = 0; t < TS_; ++t) {
        int tb = TS_ - 1 - t;
        gemm_launch(stream, hh0, H_, sw_e0_hh, gates_e, 2048,
                    g0_in + (size_t)t * B_ * 2048, 2, nullptr, 0, 0, B_, 2048, H_);
        gemm_launch(stream, hh0 + (size_t)B_ * H_, H_, sw_e0_hh + SZ_512x2048,
                    gates_e + (size_t)B_ * 2048, 2048,
                    g0_in + (size_t)TS_ * B_ * 2048 + (size_t)tb * B_ * 2048, 2,
                    nullptr, 0, 0, B_, 2048, H_);
        k_lstm_step<<<dim3(128), dim3(256), 0, stream>>>(
            gates_e, h0, c0, hh0, x1, 1024, nullptr, t, tb);
    }

    // ---- 5: encoder layer 1 input-gate precompute + recurrence ----
    for (int d = 0; d < 2; ++d)
        gemm_launch(stream, x1, 1024, sw_e1_ih + d * SZ_1024x2048,
                    g1_in + (size_t)d * TS_ * B_ * 2048, 2048,
                    e1_b + (size_t)d * 2048, 1, nullptr, 0, 0,
                    TS_ * B_, 2048, 1024);
    for (int t = 0; t < TS_; ++t) {
        int tb = TS_ - 1 - t;
        gemm_launch(stream, hh1, H_, sw_e1_hh, gates_e, 2048,
                    g1_in + (size_t)t * B_ * 2048, 2, nullptr, 0, 0, B_, 2048, H_);
        gemm_launch(stream, hh1 + (size_t)B_ * H_, H_, sw_e1_hh + SZ_512x2048,
                    gates_e + (size_t)B_ * 2048, 2048,
                    g1_in + (size_t)TS_ * B_ * 2048 + (size_t)tb * B_ * 2048, 2,
                    nullptr, 0, 0, B_, 2048, H_);
        k_lstm_step<<<dim3(128), dim3(256), 0, stream>>>(
            gates_e, h1, c1, hh1, nullptr, 0, enc_out, t, tb);
    }

    // ---- 6: decoder init + input-gate precompute ----
    k_init_dec<<<dim3(128), dim3(256), 0, stream>>>(h0, c0, dec_h, dec_c, dec_hh);
    gemm_launch(stream, x_trg, E_, sw_d_ih, gdec_in, 4096,
                d_b, 1, nullptr, 0, 0, STEPS * B_, 4096, E_);

    // ---- 7: decoder steps ----
    for (int s = 0; s < STEPS; ++s) {
        gemm_launch(stream, dec_hh, 1024, sw_d_hh, gates_d, 4096,
                    gdec_in + (size_t)s * B_ * 4096, 2, nullptr, 0, 0, B_, 4096, 1024);
        k_dec_cell<<<dim3(128), dim3(256), 0, stream>>>(gates_d, trg, s, dec_h, dec_c, dec_hh);
        k_attn<<<dim3(B_), dim3(256), 0, stream>>>(enc_out, dec_h, src, concat);
        gemm_launch(stream, concat, 2048, sw_o1, zt, 512,
                    nullptr, 0, zt_h, 512, 1, B_, 512, 2048);
        gemm_launch(stream, zt_h, 512, sw_o2, logits, V_,
                    nullptr, 0, nullptr, 0, 0, B_, V_, 512);
        k_loss<<<dim3(B_), dim3(256), 0, stream>>>(logits, trg, s, lacc, out + 1);
    }

    // ---- 8: finalize loss / denom ----
    k_finalize<<<dim3(1), dim3(256), 0, stream>>>(lacc, trg, out);
}
